// DecoderLayer_32049045963241
// MI455X (gfx1250) — compile-verified, matrix operand order checked by reading, not run on hardware
//
#include <hip/hip_runtime.h>
#include <hip/hip_bf16.h>
#include <math.h>

// ---------------- model constants ----------------
#define NB   4
#define LL   1024
#define DM   768
#define HH   12
#define DPH  64
#define DFF  3072
#define MM   (NB*LL)                    // 4096 rows
#define ATT_SCALE 0.036084391824351615f // 1/sqrt(768) (reference uses sqrt(D))

typedef __attribute__((ext_vector_type(16))) __bf16 v16bf;
typedef __attribute__((ext_vector_type(8)))  float  v8f;
typedef __attribute__((ext_vector_type(8)))  int    v8i;
typedef __attribute__((ext_vector_type(4)))  int    v4i;

// ---------------- bf16 helpers ----------------
__device__ __forceinline__ unsigned short f2bf(float f) {
  unsigned int u = __builtin_bit_cast(unsigned int, f);
  u += 0x7fffu + ((u >> 16) & 1u);          // round-to-nearest-even
  return (unsigned short)(u >> 16);
}
__device__ __forceinline__ unsigned int pack2(unsigned short lo, unsigned short hi) {
  return (unsigned int)lo | ((unsigned int)hi << 16);
}

// ---------------- async global->LDS (gfx1250 path with fallback) ----------------
__device__ __forceinline__ void async_copy16(const void* gp, void* lp) {
#if __has_builtin(__builtin_amdgcn_global_load_async_to_lds_b128)
  __builtin_amdgcn_global_load_async_to_lds_b128(
      (__attribute__((address_space(1))) v4i*)(void*)gp,
      (__attribute__((address_space(3))) v4i*)lp, 0, 0);
#else
  *(float4*)lp = *(const float4*)gp;
#endif
}
__device__ __forceinline__ void async_wait_all() {
#if __has_builtin(__builtin_amdgcn_global_load_async_to_lds_b128)
# if __has_builtin(__builtin_amdgcn_s_wait_asynccnt)
  __builtin_amdgcn_s_wait_asynccnt(0);
# else
  asm volatile("s_wait_asynccnt 0" ::: "memory");
# endif
#endif
}

// ---------------- elementwise kernels ----------------
__global__ void cvt_bf16_kernel(const float* __restrict__ in,
                                unsigned short* __restrict__ out, int n) {
  int i = blockIdx.x * blockDim.x + threadIdx.x;
  if (i < n) out[i] = f2bf(in[i]);
}

__global__ void mask_logits_kernel(const int* __restrict__ mask,
                                   float* __restrict__ out, int n) {
  int i = blockIdx.x * blockDim.x + threadIdx.x;
  if (i < n) out[i] = (1.0f - (float)mask[i]) * -10000.0f;
}

// ---------------- fused residual + LayerNorm (f32 out + optional bf16 out) ----
__global__ __launch_bounds__(256)
void add_ln_kernel(const float* __restrict__ X, const float* __restrict__ Dl,
                   const float* __restrict__ gw, const float* __restrict__ bw,
                   float* __restrict__ Yf, unsigned short* __restrict__ Yb) {
  __shared__ float red[256];
  const int row = blockIdx.x, tid = threadIdx.x;
  float v[3];
  float ls = 0.0f;
#pragma unroll
  for (int j = 0; j < 3; ++j) {
    int c = tid + j * 256;
    v[j] = X[(size_t)row * DM + c] + Dl[(size_t)row * DM + c];
    ls += v[j];
  }
  red[tid] = ls; __syncthreads();
  for (int s = 128; s > 0; s >>= 1) { if (tid < s) red[tid] += red[tid + s]; __syncthreads(); }
  const float mu = red[0] * (1.0f / DM);
  __syncthreads();
  float vs = 0.0f;
#pragma unroll
  for (int j = 0; j < 3; ++j) { float d = v[j] - mu; vs += d * d; }
  red[tid] = vs; __syncthreads();
  for (int s = 128; s > 0; s >>= 1) { if (tid < s) red[tid] += red[tid + s]; __syncthreads(); }
  const float rstd = rsqrtf(red[0] * (1.0f / DM) + 1e-5f);
#pragma unroll
  for (int j = 0; j < 3; ++j) {
    int c = tid + j * 256;
    float y = (v[j] - mu) * rstd * gw[c] + bw[c];
    Yf[(size_t)row * DM + c] = y;
    if (Yb) Yb[(size_t)row * DM + c] = f2bf(y);
  }
}

// ---------------- WMMA GEMM: C = act(A[MxK] * B[KxN] + bias) ----------------
// A,B bf16 row-major. BM=128 BN=64 BK=32, 256 threads = 8 waves,
// wave w owns rows [w*16, w*16+16) x all 64 cols -> 4 v8f accumulators.
// Fragments are built as int8 vectors (one packed dword per VGPR) and
// bit_cast to v16bf so codegen avoids per-half v_mov_b16 shuffles.
template<bool GELU, bool OUTBF>
__global__ __launch_bounds__(256)
void gemm_bf16_wmma(const unsigned short* __restrict__ A,
                    const unsigned short* __restrict__ B,
                    const float* __restrict__ bias,
                    float* __restrict__ Cf, unsigned short* __restrict__ Cb,
                    int M, int N, int K) {
  __shared__ unsigned short sA[128 * 32];
  __shared__ unsigned short sB[32 * 64];
  const int tid = threadIdx.x;
  const int wave = tid >> 5, lane = tid & 31;
  const int g = lane >> 4, n16 = lane & 15;
  const int row0 = blockIdx.y * 128, col0 = blockIdx.x * 64;

  const v8f vz = {0.f,0.f,0.f,0.f,0.f,0.f,0.f,0.f};
  v8f acc[4];
#pragma unroll
  for (int i = 0; i < 4; ++i) acc[i] = vz;

  for (int kt = 0; kt < K; kt += 32) {
    // stage A tile (128x32 bf16 = 8KB): 512 x 16B, 2 per thread
#pragma unroll
    for (int i = 0; i < 2; ++i) {
      int t = tid + i * 256;
      int r = t >> 2, c = (t & 3) * 8;
      async_copy16(A + (size_t)(row0 + r) * K + kt + c, &sA[r * 32 + c]);
    }
    { // stage B tile (32x64 bf16 = 4KB): 256 x 16B, 1 per thread
      int r = tid >> 3, c = (tid & 7) * 8;
      async_copy16(B + (size_t)(kt + r) * N + col0 + c, &sB[r * 64 + c]);
    }
    async_wait_all();
    __syncthreads();

    // A fragment (16x32) for this wave's rows: lane m = n16, K pairs per VGPR
    v8i ai;
    const int ar = wave * 16 + n16;
#pragma unroll
    for (int v = 0; v < 8; ++v) {
      int kk = ((v & 3) << 1) + (g << 3) + ((v >> 2) << 4);
      ai[v] = *(const int*)&sA[ar * 32 + kk];
    }
    const v16bf af = __builtin_bit_cast(v16bf, ai);
#pragma unroll
    for (int nt = 0; nt < 4; ++nt) {
      v8i bi;
      const int bn = nt * 16 + n16;
#pragma unroll
      for (int v = 0; v < 8; ++v) {
        int kk = ((v & 3) << 1) + (g << 3) + ((v >> 2) << 4);
        bi[v] = (int)pack2(sB[kk * 64 + bn], sB[(kk + 1) * 64 + bn]);
      }
      acc[nt] = __builtin_amdgcn_wmma_f32_16x16x32_bf16(
          false, af, false, __builtin_bit_cast(v16bf, bi),
          (short)0, acc[nt], false, false);
    }
    __syncthreads();
  }

  // epilogue: C rows = r + 8g, cols = nt*16 + n16
#pragma unroll
  for (int nt = 0; nt < 4; ++nt) {
    const int col = col0 + nt * 16 + n16;
    const float bv = bias ? bias[col] : 0.0f;
#pragma unroll
    for (int r = 0; r < 8; ++r) {
      int row = row0 + wave * 16 + r + 8 * g;
      float x = acc[nt][r] + bv;
      if (GELU) x = 0.5f * x * (1.0f + erff(x * 0.70710678118654752f));
      if (OUTBF) Cb[(size_t)row * N + col] = f2bf(x);
      else       Cf[(size_t)row * N + col] = x;
    }
  }
}

// ---------------- flash attention (one wave per (n,h,16-query tile)) --------
// Computes S^T = K * Q^T so P lands in A-fragment layout for P*V (no lane
// transpose). Masks are additive -10000, matching the reference.
__global__ __launch_bounds__(256)
void attn_wmma(const unsigned short* __restrict__ Qm,
               const unsigned short* __restrict__ Km,
               const unsigned short* __restrict__ Vm,
               const float* __restrict__ logitAdd,
               unsigned short* __restrict__ Om,
               int Lk, int causal) {
  const int tid = threadIdx.x;
  const int wave = tid >> 5, lane = tid & 31;
  const int g = lane >> 4, n16 = lane & 15;
  const int qtiles = LL / 16;
  const int wi = blockIdx.x * 8 + wave;
  const int n   = wi / (HH * qtiles);
  const int rem = wi % (HH * qtiles);
  const int h   = rem / qtiles;
  const int q0  = (rem % qtiles) * 16;

  const unsigned short* qb = Qm + ((size_t)n * LL + q0) * DM + h * DPH;
  const unsigned short* kb = Km + (size_t)n * Lk * DM + h * DPH;
  const unsigned short* vb = Vm + (size_t)n * Lk * DM + h * DPH;
  const float* la = logitAdd + (size_t)n * Lk;

  // Q^T as two B fragments (dph 0-31, 32-63); lane col = query = n16
  v16bf qf[2];
#pragma unroll
  for (int hf = 0; hf < 2; ++hf) {
    v8i qi;
#pragma unroll
    for (int v = 0; v < 8; ++v) {
      int kk = ((v & 3) << 1) + (g << 3) + ((v >> 2) << 4) + hf * 32;
      qi[v] = *(const int*)(qb + (size_t)n16 * DM + kk);
    }
    qf[hf] = __builtin_bit_cast(v16bf, qi);
  }

  const v8f vz = {0.f,0.f,0.f,0.f,0.f,0.f,0.f,0.f};
  v8f oacc[4];
#pragma unroll
  for (int i = 0; i < 4; ++i) oacc[i] = vz;
  float mrun = -3.0e38f, srun = 0.0f;
  const int q_abs = q0 + n16;
  const int kend = causal ? (q0 + 16) : Lk;

  for (int kc = 0; kc < kend; kc += 32) {
    v8f sblk[2];
#pragma unroll
    for (int blk = 0; blk < 2; ++blk) {
      const int key16 = kc + blk * 16;
      v8f sa = vz;
#pragma unroll
      for (int hf = 0; hf < 2; ++hf) {
        v8i ki;  // A fragment: 16 keys x 32 dph; lane row = key16 + n16
#pragma unroll
        for (int v = 0; v < 8; ++v) {
          int kk = ((v & 3) << 1) + (g << 3) + ((v >> 2) << 4) + hf * 32;
          ki[v] = *(const int*)(kb + (size_t)(key16 + n16) * DM + kk);
        }
        sa = __builtin_amdgcn_wmma_f32_16x16x32_bf16(
            false, __builtin_bit_cast(v16bf, ki), false, qf[hf],
            (short)0, sa, false, false);
      }
      // S^T layout: row = key = key16 + r + 8g, col = query = n16
#pragma unroll
      for (int r = 0; r < 8; ++r) {
        int key = key16 + r + 8 * g;
        float lv = la[key & (LL - 1)];
        float s = sa[r] * ATT_SCALE;
        s += (causal && key > q_abs) ? -10000.0f : lv;
        sa[r] = s;
      }
      sblk[blk] = sa;
    }

    // softmax over this 32-key chunk; per-query stats live in lanes q, q+16
    float mx = -3.0e38f;
#pragma unroll
    for (int blk = 0; blk < 2; ++blk)
#pragma unroll
      for (int r = 0; r < 8; ++r) mx = fmaxf(mx, sblk[blk][r]);
    mx = fmaxf(mx, __shfl_xor(mx, 16, 32));
    const float mnew = fmaxf(mrun, mx);
    const float alpha = __expf(mrun - mnew);
    float lsum = 0.0f;
#pragma unroll
    for (int blk = 0; blk < 2; ++blk)
#pragma unroll
      for (int r = 0; r < 8; ++r) {
        float p = __expf(sblk[blk][r] - mnew);
        sblk[blk][r] = p;
        lsum += p;
      }
    lsum += __shfl_xor(lsum, 16, 32);
    srun = srun * alpha + lsum;
    mrun = mnew;

    // rescale O accumulators: row q = r + 8g needs alpha held by lane r + 8g
#pragma unroll
    for (int r = 0; r < 8; ++r) {
      float af = __shfl(alpha, r + 8 * g, 32);
#pragma unroll
      for (int nt = 0; nt < 4; ++nt) oacc[nt][r] *= af;
    }

    // P as A fragment: pair (2j,2j+1)+8g+16blk == sblk[blk][2j..2j+1]
    v8i pi;
#pragma unroll
    for (int v = 0; v < 8; ++v) {
      int blk = v >> 2, j = v & 3;
      pi[v] = (int)pack2(f2bf(sblk[blk][2 * j]), f2bf(sblk[blk][2 * j + 1]));
    }
    const v16bf pf = __builtin_bit_cast(v16bf, pi);
    // V chunk (32 keys x 64 dph) as four 32x16 B fragments
#pragma unroll
    for (int nt = 0; nt < 4; ++nt) {
      v8i vi;
#pragma unroll
      for (int v = 0; v < 8; ++v) {
        int kk = ((v & 3) << 1) + (g << 3) + ((v >> 2) << 4);
        vi[v] = (int)pack2(vb[(size_t)(kc + kk) * DM + nt * 16 + n16],
                           vb[(size_t)(kc + kk + 1) * DM + nt * 16 + n16]);
      }
      oacc[nt] = __builtin_amdgcn_wmma_f32_16x16x32_bf16(
          false, pf, false, __builtin_bit_cast(v16bf, vi),
          (short)0, oacc[nt], false, false);
    }
  }

  // normalize and store (bf16, concat-heads layout)
#pragma unroll
  for (int r = 0; r < 8; ++r) {
    float sr = __shfl(srun, r + 8 * g, 32);
    float inv = (sr > 0.0f) ? 1.0f / sr : 0.0f;
    int row = q0 + r + 8 * g;
#pragma unroll
    for (int nt = 0; nt < 4; ++nt) {
      int col = h * DPH + nt * 16 + n16;
      Om[((size_t)n * LL + row) * DM + col] = f2bf(oacc[nt][r] * inv);
    }
  }
}

// ---------------- host orchestration ----------------
extern "C" void kernel_launch(void* const* d_in, const int* in_sizes, int n_in,
                              void* d_out, int out_size, void* d_ws, size_t ws_size,
                              hipStream_t stream) {
  const float* x1    = (const float*)d_in[0];
  const float* x2    = (const float*)d_in[1];
  const int*   decm  = (const int*)d_in[2];
  const int*   encm  = (const int*)d_in[3];
  const float* Wq    = (const float*)d_in[4];
  const float* Wk    = (const float*)d_in[5];
  const float* Wv    = (const float*)d_in[6];
  const float* Wo_sa = (const float*)d_in[7];
  const float* bo_sa = (const float*)d_in[8];
  const float* g_sa  = (const float*)d_in[9];
  const float* b_sa  = (const float*)d_in[10];
  const float* Wq2   = (const float*)d_in[11];
  const float* Wk2   = (const float*)d_in[12];
  const float* Wv2   = (const float*)d_in[13];
  const float* Wo_ca = (const float*)d_in[14];
  const float* bo_ca = (const float*)d_in[15];
  const float* W1    = (const float*)d_in[16];
  const float* b1    = (const float*)d_in[17];
  const float* W2    = (const float*)d_in[18];
  const float* b2    = (const float*)d_in[19];
  const float* g_ffn = (const float*)d_in[20];
  const float* b_ffn = (const float*)d_in[21];

  char* w = (char*)d_ws;
  size_t off = 0;
  auto alloc = [&](size_t bytes) -> void* {
    void* p = w + off;
    off = (off + bytes + 255) & ~(size_t)255;
    return p;
  };
  const size_t ACT_BF = (size_t)MM * DM * 2;   // 4096x768 bf16
  const size_t ACT_F  = (size_t)MM * DM * 4;   // 4096x768 f32
  const size_t WMAT   = (size_t)DM * DM * 2;   // 768x768 bf16
  const size_t WFFN   = (size_t)DM * DFF * 2;  // 768x3072 bf16

  unsigned short* x1b   = (unsigned short*)alloc(ACT_BF);
  unsigned short* x2b   = (unsigned short*)alloc(ACT_BF);
  unsigned short* Wqb   = (unsigned short*)alloc(WMAT);
  unsigned short* Wkb   = (unsigned short*)alloc(WMAT);
  unsigned short* Wvb   = (unsigned short*)alloc(WMAT);
  unsigned short* Wosab = (unsigned short*)alloc(WMAT);
  unsigned short* Wq2b  = (unsigned short*)alloc(WMAT);
  unsigned short* Wk2b  = (unsigned short*)alloc(WMAT);
  unsigned short* Wv2b  = (unsigned short*)alloc(WMAT);
  unsigned short* Wocab = (unsigned short*)alloc(WMAT);
  unsigned short* W1b   = (unsigned short*)alloc(WFFN);
  unsigned short* W2b   = (unsigned short*)alloc(WFFN);
  float* decL = (float*)alloc((size_t)NB * LL * 4);
  float* encL = (float*)alloc((size_t)NB * LL * 4);
  unsigned short* Qb  = (unsigned short*)alloc(ACT_BF);
  unsigned short* Kb  = (unsigned short*)alloc(ACT_BF);
  unsigned short* Vb  = (unsigned short*)alloc(ACT_BF);
  unsigned short* Ob  = (unsigned short*)alloc(ACT_BF);
  float* P1 = (float*)alloc(ACT_F);
  float* hF = (float*)alloc(ACT_F);
  unsigned short* hB = (unsigned short*)alloc(ACT_BF);
  unsigned short* Q2b = (unsigned short*)alloc(ACT_BF);
  unsigned short* K2b = (unsigned short*)alloc(ACT_BF);
  unsigned short* V2b = (unsigned short*)alloc(ACT_BF);
  unsigned short* O2b = (unsigned short*)alloc(ACT_BF);
  float* P2  = (float*)alloc(ACT_F);
  float* h2F = (float*)alloc(ACT_F);
  unsigned short* h2B = (unsigned short*)alloc(ACT_BF);
  unsigned short* T1b = (unsigned short*)alloc((size_t)MM * DFF * 2);
  float* T2 = (float*)alloc(ACT_F);
  (void)ws_size; (void)n_in; (void)in_sizes; (void)out_size;

  auto cvt = [&](const float* src, unsigned short* dst, int n) {
    cvt_bf16_kernel<<<(n + 255) / 256, 256, 0, stream>>>(src, dst, n);
  };
  cvt(x1, x1b, MM * DM);
  cvt(x2, x2b, MM * DM);
  cvt(Wq, Wqb, DM * DM);    cvt(Wk, Wkb, DM * DM);
  cvt(Wv, Wvb, DM * DM);    cvt(Wo_sa, Wosab, DM * DM);
  cvt(Wq2, Wq2b, DM * DM);  cvt(Wk2, Wk2b, DM * DM);
  cvt(Wv2, Wv2b, DM * DM);  cvt(Wo_ca, Wocab, DM * DM);
  cvt(W1, W1b, DM * DFF);   cvt(W2, W2b, DFF * DM);
  mask_logits_kernel<<<(NB * LL) / 256, 256, 0, stream>>>(decm, decL, NB * LL);
  mask_logits_kernel<<<(NB * LL) / 256, 256, 0, stream>>>(encm, encL, NB * LL);

  const dim3 gProj(DM / 64, MM / 128);   // (12, 32)
  const dim3 gFfn1(DFF / 64, MM / 128);  // (48, 32)
  const int attnBlocks = (NB * HH * (LL / 16)) / 8;  // 384

  // ---- self-attention ----
  gemm_bf16_wmma<false, true><<<gProj, 256, 0, stream>>>(x1b, Wqb, nullptr, nullptr, Qb, MM, DM, DM);
  gemm_bf16_wmma<false, true><<<gProj, 256, 0, stream>>>(x1b, Wkb, nullptr, nullptr, Kb, MM, DM, DM);
  gemm_bf16_wmma<false, true><<<gProj, 256, 0, stream>>>(x1b, Wvb, nullptr, nullptr, Vb, MM, DM, DM);
  attn_wmma<<<attnBlocks, 256, 0, stream>>>(Qb, Kb, Vb, decL, Ob, LL, 1);
  gemm_bf16_wmma<false, false><<<gProj, 256, 0, stream>>>(Ob, Wosab, bo_sa, P1, nullptr, MM, DM, DM);
  add_ln_kernel<<<MM, 256, 0, stream>>>(x1, P1, g_sa, b_sa, hF, hB);

  // ---- cross-attention (reuses g_sa/b_sa per the reference) ----
  gemm_bf16_wmma<false, true><<<gProj, 256, 0, stream>>>(hB,  Wq2b, nullptr, nullptr, Q2b, MM, DM, DM);
  gemm_bf16_wmma<false, true><<<gProj, 256, 0, stream>>>(x2b, Wk2b, nullptr, nullptr, K2b, MM, DM, DM);
  gemm_bf16_wmma<false, true><<<gProj, 256, 0, stream>>>(x2b, Wv2b, nullptr, nullptr, V2b, MM, DM, DM);
  attn_wmma<<<attnBlocks, 256, 0, stream>>>(Q2b, K2b, V2b, encL, O2b, LL, 0);
  gemm_bf16_wmma<false, false><<<gProj, 256, 0, stream>>>(O2b, Wocab, bo_ca, P2, nullptr, MM, DM, DM);
  add_ln_kernel<<<MM, 256, 0, stream>>>(hF, P2, g_sa, b_sa, h2F, h2B);

  // ---- FFN ----
  gemm_bf16_wmma<true, true><<<gFfn1, 256, 0, stream>>>(h2B, W1b, b1, nullptr, T1b, MM, DFF, DM);
  gemm_bf16_wmma<false, false><<<gProj, 256, 0, stream>>>(T1b, W2b, b2, T2, nullptr, MM, DM, DFF);
  add_ln_kernel<<<MM, 256, 0, stream>>>(h2F, T2, g_ffn, b_ffn, (float*)d_out, nullptr);
}
